// MPN_64132451664100
// MI455X (gfx1250) — compile-verified, matrix operand order checked by reading, not run on hardware
//
#include <hip/hip_runtime.h>

#define HID      256
#define E_BONDS  131072
#define NATOMS   65536
#define MAXNB    6
#define APM      32
#define NMOL     2048
#define KPAD_FB  160   // 157 padded
#define KPAD_AI  416   // 399 padded

typedef __attribute__((ext_vector_type(16))) __bf16 v16bf;
typedef __attribute__((ext_vector_type(8)))  float  v8f;

union Frag {
  unsigned short s[16];
  uint4          u[2];
  v16bf          v;
};

__device__ __forceinline__ unsigned short f32_to_bf16(float f) {
  unsigned int u = __float_as_uint(f);
  u += 0x7FFFu + ((u >> 16) & 1u);           // round-to-nearest-even
  return (unsigned short)(u >> 16);
}
__device__ __forceinline__ float bf16_to_f32(unsigned short h) {
  return __uint_as_float(((unsigned int)h) << 16);
}

// ---------------------------------------------------------------------------
// Swizzle an f32 weight [Ksrc,256] into bf16 WMMA B-fragment order, padded to
// [Kpad,256].  Layout: [ktile][ntile][lane][e] so each wave's B fragment is a
// single contiguous 32-byte load.
// K(e,half) = (e/8)*16 + (e%8) + 8*half  (per CDNA5 ISA 16-bit operand layout)
__global__ void swizzle_weight_kernel(const float* __restrict__ src, int Ksrc,
                                      unsigned short* __restrict__ dst, int Kpad) {
  int idx = blockIdx.x * blockDim.x + threadIdx.x;
  if (idx >= Kpad * HID) return;
  int k = idx / HID, n = idx % HID;
  float v = (k < Ksrc) ? src[(size_t)k * HID + n] : 0.0f;
  int ktile = k >> 5, kin = k & 31;
  int q = kin >> 3, r = kin & 7;
  int hf = q & 1, e = ((q >> 1) << 3) + r;
  int ntile = n >> 4, nin = n & 15;
  int lane = nin + 16 * hf;
  size_t off = (((size_t)(ktile * 16 + ntile) * 32 + lane) << 4) + e;
  dst[off] = f32_to_bf16(v);
}

// fbonds f32 [E,157] -> bf16 row-major padded [E,160]
__global__ void prep_fbonds_kernel(const float* __restrict__ src,
                                   unsigned short* __restrict__ dst) {
  long long idx = (long long)blockIdx.x * blockDim.x + threadIdx.x;
  if (idx >= (long long)E_BONDS * KPAD_FB) return;
  int row = (int)(idx / KPAD_FB), c = (int)(idx % KPAD_FB);
  dst[idx] = (c < 157) ? f32_to_bf16(src[(size_t)row * 157 + c]) : (unsigned short)0;
}

// ---------------------------------------------------------------------------
// Generic WMMA GEMM: C[M,256] = A_bf[M,Kpad] @ B_sw[Kpad,256]
// 128 threads = 4 waves; wave w -> cols [64w,64w+64); block -> rows [32b,32b+32)
// Each wave: 2 M-tiles x 4 N-tiles = 8 WMMAs per k-step from 2 A + 4 B frags.
// flags: bit0 = ReLU; bit1 = f32 output gets PRE-activation value.
__global__ __launch_bounds__(128) void gemm_wmma_kernel(
    const unsigned short* __restrict__ A, int lda,
    const unsigned short* __restrict__ Bsw, int Kpad,
    const float* __restrict__ addend, const float* __restrict__ bias,
    float* __restrict__ outF, unsigned short* __restrict__ outBF, int flags) {
  const int lane = threadIdx.x & 31;
  const int wave = threadIdx.x >> 5;
  const int hf   = lane >> 4;
  const int l16  = lane & 15;
  const int row0 = blockIdx.x * 32;
  const int nt0  = wave * 4;

  v8f zero = {0.f,0.f,0.f,0.f,0.f,0.f,0.f,0.f};
  v8f acc[2][4] = {{zero, zero, zero, zero}, {zero, zero, zero, zero}};

  const unsigned short* ap0 = A + (size_t)(row0 + l16) * lda;
  const unsigned short* ap1 = A + (size_t)(row0 + 16 + l16) * lda;
  // running pointer into swizzled B: tile (kt, nt0+j) at bp + j*512; +8192/kt
  const unsigned short* bp = Bsw + (((size_t)nt0 * 32 + lane) << 4);

  const int ktn = Kpad >> 5;
  for (int kt = 0; kt < ktn; ++kt) {
    const int base = kt << 5;
    Frag a0, a1;
    a0.u[0] = *reinterpret_cast<const uint4*>(ap0 + base + 8 * hf);
    a0.u[1] = *reinterpret_cast<const uint4*>(ap0 + base + 16 + 8 * hf);
    a1.u[0] = *reinterpret_cast<const uint4*>(ap1 + base + 8 * hf);
    a1.u[1] = *reinterpret_cast<const uint4*>(ap1 + base + 16 + 8 * hf);
    if (kt + 1 < ktn) {
      __builtin_prefetch(ap0 + base + 32, 0, 3);   // global_prefetch_b8
      __builtin_prefetch(ap1 + base + 32, 0, 3);
    }
#pragma unroll
    for (int j = 0; j < 4; ++j) {
      const v16bf b = *reinterpret_cast<const v16bf*>(bp + (size_t)j * 512);
      acc[0][j] = __builtin_amdgcn_wmma_f32_16x16x32_bf16(
          false, a0.v, false, b, (short)0, acc[0][j], false, false);
      acc[1][j] = __builtin_amdgcn_wmma_f32_16x16x32_bf16(
          false, a1.v, false, b, (short)0, acc[1][j], false, false);
    }
    bp += 16 * 32 * 16;   // advance one k-tile row of swizzled B (8192 elems)
  }

#pragma unroll
  for (int mt2 = 0; mt2 < 2; ++mt2) {
#pragma unroll
    for (int j = 0; j < 4; ++j) {
      const int col = (nt0 + j) * 16 + l16;
#pragma unroll
      for (int i = 0; i < 8; ++i) {
        const int row = row0 + mt2 * 16 + i + 8 * hf;
        const size_t idx = (size_t)row * HID + col;
        float v = acc[mt2][j][i];
        if (addend) v += addend[idx];
        if (bias)   v += bias[col];
        float act = (flags & 1) ? fmaxf(v, 0.0f) : v;
        if (outF)  outF[idx] = (flags & 2) ? v : act;
        if (outBF) outBF[idx] = f32_to_bf16(act);
      }
    }
  }
}

// ---------------------------------------------------------------------------
// Message attention: nei[e] = sum_j softmax_j(msg[bgraph[e,j]] . wma) * msg[..]
// One wave per bond; lane owns 8 contiguous hidden columns.
__global__ __launch_bounds__(256) void attmsg_kernel(
    const unsigned short* __restrict__ msg, const int* __restrict__ bgraph,
    const float* __restrict__ wma, unsigned short* __restrict__ nei_out) {
  const int lane = threadIdx.x & 31;
  const int bond = blockIdx.x * 8 + (threadIdx.x >> 5);
  const int c0 = lane * 8;

  float wv[8];
#pragma unroll
  for (int i = 0; i < 8; ++i) wv[i] = wma[c0 + i];

  float m[MAXNB][8];
  float d[MAXNB];
#pragma unroll
  for (int j = 0; j < MAXNB; ++j) {
    const int nb = bgraph[bond * MAXNB + j];
    uint4 pk = *reinterpret_cast<const uint4*>(msg + (size_t)nb * HID + c0);
    unsigned int w[4] = {pk.x, pk.y, pk.z, pk.w};
    float p = 0.0f;
#pragma unroll
    for (int i = 0; i < 8; ++i) {
      unsigned short h = (unsigned short)((w[i >> 1] >> ((i & 1) * 16)) & 0xFFFFu);
      m[j][i] = bf16_to_f32(h);
      p += m[j][i] * wv[i];
    }
#pragma unroll
    for (int off = 16; off >= 1; off >>= 1) p += __shfl_xor(p, off, 32);
    d[j] = p;
  }
  float mx = d[0];
#pragma unroll
  for (int j = 1; j < MAXNB; ++j) mx = fmaxf(mx, d[j]);
  float a[MAXNB], s = 0.0f;
#pragma unroll
  for (int j = 0; j < MAXNB; ++j) { a[j] = __expf(d[j] - mx); s += a[j]; }
  const float inv = 1.0f / s;

  unsigned int out[4];
#pragma unroll
  for (int p2 = 0; p2 < 4; ++p2) {
    float o0 = 0.f, o1 = 0.f;
#pragma unroll
    for (int j = 0; j < MAXNB; ++j) {
      o0 += a[j] * m[j][2 * p2];
      o1 += a[j] * m[j][2 * p2 + 1];
    }
    unsigned short lo  = f32_to_bf16(o0 * inv);
    unsigned short hi2 = f32_to_bf16(o1 * inv);
    out[p2] = (unsigned int)lo | ((unsigned int)hi2 << 16);
  }
  uint4 st = {out[0], out[1], out[2], out[3]};
  *reinterpret_cast<uint4*>(nei_out + (size_t)bond * HID + c0) = st;
}

// ---------------------------------------------------------------------------
// nei_a = sum_j msg[agraph[a,j]] ; ainput = concat(fatoms, nei_a) padded to 416
__global__ __launch_bounds__(256) void atomgather_kernel(
    const float* __restrict__ fatoms, const unsigned short* __restrict__ msg,
    const int* __restrict__ agraph, unsigned short* __restrict__ ainput) {
  const int lane = threadIdx.x & 31;
  const int atom = blockIdx.x * 8 + (threadIdx.x >> 5);
  unsigned short* arow = ainput + (size_t)atom * KPAD_AI;

  for (int c = lane; c < 143; c += 32)
    arow[c] = f32_to_bf16(fatoms[(size_t)atom * 143 + c]);

  const int c0 = lane * 8;
  float acc[8] = {0.f,0.f,0.f,0.f,0.f,0.f,0.f,0.f};
#pragma unroll
  for (int j = 0; j < MAXNB; ++j) {
    const int nb = agraph[atom * MAXNB + j];
    uint4 pk = *reinterpret_cast<const uint4*>(msg + (size_t)nb * HID + c0);
    unsigned int w[4] = {pk.x, pk.y, pk.z, pk.w};
#pragma unroll
    for (int i = 0; i < 8; ++i)
      acc[i] += bf16_to_f32((unsigned short)((w[i >> 1] >> ((i & 1) * 16)) & 0xFFFFu));
  }
#pragma unroll
  for (int i = 0; i < 8; ++i)
    arow[143 + c0 + i] = f32_to_bf16(acc[i]);

  if (lane < KPAD_AI - 399) arow[399 + lane] = 0;
}

// ---------------------------------------------------------------------------
// Per-molecule attention: scores = (hW_a) h^T ; softmax ; attnh = att_w h
// One 128-thread block (4 waves) per molecule; tiles via WMMA from LDS.
__global__ __launch_bounds__(128) void molattn_kernel(
    const unsigned short* __restrict__ h_bf,
    const unsigned short* __restrict__ hwa_bf,
    unsigned short* __restrict__ attnh_bf) {
  __shared__ __align__(16) unsigned short h_s[APM * HID];
  __shared__ __align__(16) unsigned short w_s[APM * HID];
  __shared__ float sc_s[APM * APM];
  __shared__ __align__(16) unsigned short aw_s[APM * APM];

  const int mol = blockIdx.x;
  const int tid = threadIdx.x;
  const int lane = tid & 31, wave = tid >> 5;
  const int hf = lane >> 4, l16 = lane & 15;

  const uint4* gh = reinterpret_cast<const uint4*>(h_bf  + (size_t)mol * APM * HID);
  const uint4* gw = reinterpret_cast<const uint4*>(hwa_bf + (size_t)mol * APM * HID);
  uint4* sh = reinterpret_cast<uint4*>(h_s);
  uint4* sw = reinterpret_cast<uint4*>(w_s);
  for (int t = tid; t < APM * HID / 8; t += 128) { sh[t] = gh[t]; sw[t] = gw[t]; }
  __syncthreads();

  { // scores: 2x2 tiles of 16x16, one per wave
    const int mt = wave >> 1, nt = wave & 1;
    v8f acc = {0.f,0.f,0.f,0.f,0.f,0.f,0.f,0.f};
    const int ar = mt * 16 + l16;       // hWa row
    const int br = nt * 16 + l16;       // h row (== score column)
#pragma unroll
    for (int kt = 0; kt < 8; ++kt) {
      const int base = kt * 32;
      Frag a, b;
      a.u[0] = *reinterpret_cast<const uint4*>(w_s + ar * HID + base + 8 * hf);
      a.u[1] = *reinterpret_cast<const uint4*>(w_s + ar * HID + base + 16 + 8 * hf);
      b.u[0] = *reinterpret_cast<const uint4*>(h_s + br * HID + base + 8 * hf);
      b.u[1] = *reinterpret_cast<const uint4*>(h_s + br * HID + base + 16 + 8 * hf);
      acc = __builtin_amdgcn_wmma_f32_16x16x32_bf16(
          false, a.v, false, b.v, (short)0, acc, false, false);
    }
#pragma unroll
    for (int i = 0; i < 8; ++i)
      sc_s[(mt * 16 + i + 8 * hf) * APM + nt * 16 + l16] = acc[i];
  }
  __syncthreads();

  if (tid < APM) { // softmax: lane = row
    float vals[APM];
    float mx = -1e30f;
#pragma unroll
    for (int c = 0; c < APM; ++c) { vals[c] = sc_s[tid * APM + c]; mx = fmaxf(mx, vals[c]); }
    float s = 0.0f;
#pragma unroll
    for (int c = 0; c < APM; ++c) { vals[c] = __expf(vals[c] - mx); s += vals[c]; }
    const float inv = 1.0f / s;
#pragma unroll
    for (int c = 0; c < APM; ++c) aw_s[tid * APM + c] = f32_to_bf16(vals[c] * inv);
  }
  __syncthreads();

  // attnh = att_w[32,32] @ h[32,256]: 32 tiles over 4 waves
  for (int t = wave; t < 32; t += 4) {
    const int mt = t >> 4, nt = t & 15;
    Frag a, b;
    const int ar = mt * 16 + l16;
    a.u[0] = *reinterpret_cast<const uint4*>(aw_s + ar * APM + 8 * hf);
    a.u[1] = *reinterpret_cast<const uint4*>(aw_s + ar * APM + 16 + 8 * hf);
    const int bc = nt * 16 + l16;
#pragma unroll
    for (int e = 0; e < 16; ++e) {
      const int K = ((e >> 3) << 4) + (e & 7) + 8 * hf;
      b.s[e] = h_s[K * HID + bc];
    }
    v8f acc = {0.f,0.f,0.f,0.f,0.f,0.f,0.f,0.f};
    acc = __builtin_amdgcn_wmma_f32_16x16x32_bf16(
        false, a.v, false, b.v, (short)0, acc, false, false);
#pragma unroll
    for (int i = 0; i < 8; ++i) {
      const int row = mol * APM + mt * 16 + i + 8 * hf;
      attnh_bf[(size_t)row * HID + nt * 16 + l16] = f32_to_bf16(acc[i]);
    }
  }
}

// mol_vecs = (atom_h + att_h).sum over 32 atoms / 32
__global__ __launch_bounds__(256) void pool_kernel(
    const float* __restrict__ atomh, const float* __restrict__ atth,
    float* __restrict__ out) {
  const int mol = blockIdx.x, c = threadIdx.x;
  float s = 0.0f;
  for (int a2 = 0; a2 < APM; ++a2) {
    const size_t idx = (size_t)(mol * APM + a2) * HID + c;
    s += atomh[idx] + atth[idx];
  }
  out[(size_t)mol * HID + c] = s * (1.0f / APM);
}

// ---------------------------------------------------------------------------
extern "C" void kernel_launch(void* const* d_in, const int* in_sizes, int n_in,
                              void* d_out, int out_size, void* d_ws, size_t ws_size,
                              hipStream_t stream) {
  (void)in_sizes; (void)n_in; (void)out_size; (void)ws_size;
  const float* fatoms = (const float*)d_in[0];
  const float* fbonds = (const float*)d_in[1];
  const int*   agraph = (const int*)d_in[2];
  const int*   bgraph = (const int*)d_in[3];
  const float* W_i    = (const float*)d_in[5];
  const float* W_h    = (const float*)d_in[6];
  const float* W_o_w  = (const float*)d_in[7];
  const float* W_o_b  = (const float*)d_in[8];
  const float* W_a    = (const float*)d_in[9];
  const float* W_b_w  = (const float*)d_in[10];
  const float* W_b_b  = (const float*)d_in[11];
  const float* W_ma1  = (const float*)d_in[12];
  float* out = (float*)d_out;

  char* ws = (char*)d_ws;
  size_t off = 0;
  auto alloc = [&](size_t bytes) { size_t o = off; off = (off + bytes + 255) & ~(size_t)255; return o; };

  unsigned short* Wi_sw   = (unsigned short*)(ws + alloc((size_t)KPAD_FB * HID * 2));
  unsigned short* Wh_sw   = (unsigned short*)(ws + alloc((size_t)HID * HID * 2));
  unsigned short* Wo_sw   = (unsigned short*)(ws + alloc((size_t)KPAD_AI * HID * 2));
  unsigned short* Wa_sw   = (unsigned short*)(ws + alloc((size_t)HID * HID * 2));
  unsigned short* Wb_sw   = (unsigned short*)(ws + alloc((size_t)HID * HID * 2));
  unsigned short* fb_bf   = (unsigned short*)(ws + alloc((size_t)E_BONDS * KPAD_FB * 2));
  float*          binput  = (float*)         (ws + alloc((size_t)E_BONDS * HID * 4));
  unsigned short* msg_bf  = (unsigned short*)(ws + alloc((size_t)E_BONDS * HID * 2));
  unsigned short* nei_bf  = (unsigned short*)(ws + alloc((size_t)E_BONDS * HID * 2));
  unsigned short* ain_bf  = (unsigned short*)(ws + alloc((size_t)NATOMS * KPAD_AI * 2));
  float*          atomh   = (float*)         (ws + alloc((size_t)NATOMS * HID * 4));
  unsigned short* h_bf    = (unsigned short*)(ws + alloc((size_t)NATOMS * HID * 2));
  unsigned short* hwa_bf  = (unsigned short*)(ws + alloc((size_t)NATOMS * HID * 2));
  unsigned short* attnh   = (unsigned short*)(ws + alloc((size_t)NATOMS * HID * 2));
  float*          atth    = (float*)         (ws + alloc((size_t)NATOMS * HID * 4));

  // 1. weight prep (bf16 + B-fragment swizzle)
  swizzle_weight_kernel<<<(KPAD_FB * HID + 255) / 256, 256, 0, stream>>>(W_i, 157, Wi_sw, KPAD_FB);
  swizzle_weight_kernel<<<(HID * HID + 255) / 256, 256, 0, stream>>>(W_h, HID, Wh_sw, HID);
  swizzle_weight_kernel<<<(KPAD_AI * HID + 255) / 256, 256, 0, stream>>>(W_o_w, 399, Wo_sw, KPAD_AI);
  swizzle_weight_kernel<<<(HID * HID + 255) / 256, 256, 0, stream>>>(W_a, HID, Wa_sw, HID);
  swizzle_weight_kernel<<<(HID * HID + 255) / 256, 256, 0, stream>>>(W_b_w, HID, Wb_sw, HID);
  prep_fbonds_kernel<<<(E_BONDS * KPAD_FB + 255) / 256, 256, 0, stream>>>(fbonds, fb_bf);

  // 2. binput = fbonds@W_i (f32 pre-act kept), message = relu(binput) (bf16)
  gemm_wmma_kernel<<<E_BONDS / 32, 128, 0, stream>>>(
      fb_bf, KPAD_FB, Wi_sw, KPAD_FB, nullptr, nullptr, binput, msg_bf, /*relu|preact*/3);

  // 3. DEPTH-1 message-attention updates
  for (int it = 0; it < 3; ++it) {
    attmsg_kernel<<<E_BONDS / 8, 256, 0, stream>>>(msg_bf, bgraph, W_ma1, nei_bf);
    gemm_wmma_kernel<<<E_BONDS / 32, 128, 0, stream>>>(
        nei_bf, HID, Wh_sw, HID, binput, nullptr, nullptr, msg_bf, /*relu*/1);
  }

  // 4. atom readout: ainput = [fatoms | sum msg[agraph]]
  atomgather_kernel<<<NATOMS / 8, 256, 0, stream>>>(fatoms, msg_bf, agraph, ain_bf);
  gemm_wmma_kernel<<<NATOMS / 32, 128, 0, stream>>>(
      ain_bf, KPAD_AI, Wo_sw, KPAD_AI, nullptr, W_o_b, atomh, h_bf, /*relu*/1);

  // 5. molecule self-attention pooling
  gemm_wmma_kernel<<<NATOMS / 32, 128, 0, stream>>>(
      h_bf, HID, Wa_sw, HID, nullptr, nullptr, nullptr, hwa_bf, /*none*/0);
  molattn_kernel<<<NMOL, 128, 0, stream>>>(h_bf, hwa_bf, attnh);
  gemm_wmma_kernel<<<NATOMS / 32, 128, 0, stream>>>(
      attnh, HID, Wb_sw, HID, nullptr, W_b_b, atth, nullptr, /*relu*/1);

  // 6. mean pool
  pool_kernel<<<NMOL, 256, 0, stream>>>(atomh, atth, out);
}
